// EncoderCoreDecoder_40939628265671
// MI455X (gfx1250) — matrix operand order repaired and checked by reading, not compiled, not run on hardware
//
#include <hip/hip_runtime.h>

// ---------------------------------------------------------------------------
// GraphNet encode-process-decode forward for MI455X (gfx1250, wave32).
// All GEMMs (N=128) run as bf16 WMMA v_wmma_f32_16x16x32_bf16 with f32 accum.
// Barrier-free GEMM: each lane loads its A row segments directly (2x b128),
// weights pre-swizzled into a per-lane B VGPR layout in contiguous k order (k = 16*half + s) (2x b128),
// 4 waves/WG x 2 column-tiles/wave (each A load feeds two WMMAs).
// Gathers materialize concatenated operands in bf16 (halves HBM traffic).
// ---------------------------------------------------------------------------

typedef __attribute__((ext_vector_type(16))) __bf16        bf16x16;
typedef __attribute__((ext_vector_type(8)))  float         f32x8;
typedef __attribute__((ext_vector_type(8)))  unsigned int  u32x8;

static __device__ __forceinline__ unsigned short f2bf(float f) {
  union { float f; unsigned u; } c; c.f = f;
  unsigned u = c.u;
  return (unsigned short)((u + 0x7FFFu + ((u >> 16) & 1u)) >> 16); // RNE
}
static __device__ __forceinline__ unsigned pack2(float lo, float hi) {
  return (unsigned)f2bf(lo) | ((unsigned)f2bf(hi) << 16);
}

// --- weight swizzle: f32 W[Kin x 128] -> packed bf16 pairs in the per-lane
//     B-matrix order: Wsw[((chunk*128 + col)*2 + lanehalf)*8 + v]
//     holds K = chunk*32 + lanehalf*16 + {2v, 2v+1}; zero-padded to Kpad.
__global__ void swizzle_weights(const float* __restrict__ W,
                                unsigned* __restrict__ Wsw,
                                int Kin, int total) {
  int t = blockIdx.x * blockDim.x + threadIdx.x;
  if (t >= total) return;
  int v    = t & 7;
  int half = (t >> 3) & 1;
  int col  = (t >> 4) & 127;
  int c    = t >> 11;
  int k0   = c * 32 + half * 16 + 2 * v;
  float lo = (k0     < Kin) ? W[(size_t)k0 * 128 + col]       : 0.f;
  float hi = (k0 + 1 < Kin) ? W[(size_t)(k0 + 1) * 128 + col] : 0.f;
  Wsw[t] = pack2(lo, hi);
}

// --- generic GEMM: C[M,128] = act(A[M,K](bf16) @ W[K,128] + bias)
//     K multiple of 32. 4 waves/WG; wave w owns col tiles {16w, 16w+64}.
//     A layout (contiguous k order, matching Wsw): lane l holds row (l&15); lane-half (l>>4)
//     selects the K sub-offset, so each chunk's A regs are two contiguous
//     16-byte row segments -> direct global b128 loads, no LDS, no barriers.
__global__ void __launch_bounds__(128)
gemm_bf16_n128(const unsigned short* __restrict__ Ah,
               const unsigned* __restrict__ Wsw,
               const float* __restrict__ bias,
               float* __restrict__ outF,            // optional f32 out
               unsigned short* __restrict__ outH,   // optional bf16 out
               int M, int K, int relu) {
  const int wave = threadIdx.x >> 5;                // 0..3
  const int lane = threadIdx.x & 31;
  const int half = lane >> 4;
  const int n16  = lane & 15;
  const int row0 = blockIdx.x << 4;
  const int col0 = (wave << 4) + n16;               // tile 0 column
  const int col1 = col0 + 64;                       // tile 1 column
  const int Ku   = K >> 1;                          // uints per A row
  int am = row0 + n16; if (am >= M) am = M - 1;     // clamp (M%16==0 here)
  const unsigned* Arow = (const unsigned*)Ah + (size_t)am * Ku;
  f32x8 acc0 = {0.f,0.f,0.f,0.f,0.f,0.f,0.f,0.f};
  f32x8 acc1 = {0.f,0.f,0.f,0.f,0.f,0.f,0.f,0.f};
  const int nch = K >> 5;
  for (int c = 0; c < nch; ++c) {
    union { u32x8 u; bf16x16 h; } a, b0, b1;
    // A: K = half*16 + {0..15} within this chunk
    const unsigned* ap = Arow + (c << 4) + (half << 3);
#pragma unroll
    for (int v = 0; v < 4; ++v) { a.u[v] = ap[v]; a.u[4 + v] = ap[4 + v]; }
    const unsigned* wp0 = Wsw + ((((size_t)c * 128 + col0) * 2 + half) << 3);
    const unsigned* wp1 = Wsw + ((((size_t)c * 128 + col1) * 2 + half) << 3);
#pragma unroll
    for (int v = 0; v < 8; ++v) { b0.u[v] = wp0[v]; b1.u[v] = wp1[v]; }
    if (c + 1 < nch)
      __builtin_prefetch(Arow + ((c + 1) << 4), 0, 1);  // global_prefetch_b8
    acc0 = __builtin_amdgcn_wmma_f32_16x16x32_bf16(false, a.h, false, b0.h,
                                                   (short)0, acc0, false, false);
    acc1 = __builtin_amdgcn_wmma_f32_16x16x32_bf16(false, a.h, false, b1.h,
                                                   (short)0, acc1, false, false);
  }
  float bn0 = bias ? bias[col0] : 0.f;
  float bn1 = bias ? bias[col1] : 0.f;
#pragma unroll
  for (int r = 0; r < 8; ++r) {
    int gr = row0 + r + (half << 3);                // C layout: M = r + 8*lanehalf
    if (gr < M) {
      float v0 = acc0[r] + bn0, v1 = acc1[r] + bn1;
      if (relu) { v0 = fmaxf(v0, 0.f); v1 = fmaxf(v1, 0.f); }
      size_t o = (size_t)gr * 128;
      if (outF) { outF[o + col0] = v0;      outF[o + col1] = v1; }
      if (outH) { outH[o + col0] = f2bf(v0); outH[o + col1] = f2bf(v1); }
    }
  }
}

// --- LayerNorm over rows of 128 (ReLU already applied upstream), in place.
__global__ void __launch_bounds__(256)
ln_rows(float* __restrict__ X, const float* __restrict__ g,
        const float* __restrict__ b, int M) {
  int row  = (blockIdx.x << 3) + (threadIdx.x >> 5);
  int lane = threadIdx.x & 31;
  if (row >= M) return;
  float* p = X + (size_t)row * 128 + lane * 4;
  float4 x = *(const float4*)p;
  float s = x.x + x.y + x.z + x.w;
  float q = x.x*x.x + x.y*x.y + x.z*x.z + x.w*x.w;
#pragma unroll
  for (int m = 16; m; m >>= 1) { s += __shfl_xor(s, m); q += __shfl_xor(q, m); }
  float mu  = s * 0.0078125f;
  float var = q * 0.0078125f - mu * mu;
  float inv = rsqrtf(var + 1e-5f);
  float4 gg = *(const float4*)(g + lane * 4);
  float4 bb = *(const float4*)(b + lane * 4);
  x.x = (x.x - mu) * inv * gg.x + bb.x;
  x.y = (x.y - mu) * inv * gg.y + bb.y;
  x.z = (x.z - mu) * inv * gg.z + bb.z;
  x.w = (x.w - mu) * inv * gg.w + bb.w;
  *(float4*)p = x;
}

// --- encoder input slices -> bf16, K padded to 32
__global__ void gather_xenc(const float* __restrict__ x,
                            unsigned short* __restrict__ xv,
                            unsigned short* __restrict__ xc, int n_v) {
  int t = blockIdx.x * blockDim.x + threadIdx.x;
  if (t >= n_v * 32) return;
  int i = t >> 5, k = t & 31;
  float f = x[(size_t)i * 11 + 1 + k];              // x[:,1+k] (k<10 valid)
  xv[t] = f2bf(k < 10 ? f : 0.f);
  xc[t] = f2bf(k <  6 ? f : 0.f);
}

// --- h = where(type==0, enc_v, 0) + where(type==1, enc_c, 0)
__global__ void select_h(const float* __restrict__ x,
                         const float* __restrict__ ev,
                         const float* __restrict__ ec,
                         float* __restrict__ h, int n_v) {
  int t = blockIdx.x * blockDim.x + threadIdx.x;
  if (t >= n_v * 128) return;
  float ty = x[(size_t)(t >> 7) * 11];
  h[t] = (ty == 0.f) ? ev[t] : ((ty == 1.f) ? ec[t] : 0.f);
}

// --- e_cat = [xin[row](256) | xin[col](256) | edge_attr(2) | le(128) |
//              u[eidx](4) | lu[eidx](128)] padded to 800, bf16
__global__ void gather_ecat(const float* __restrict__ h, const float* __restrict__ lv,
                            const float* __restrict__ le, const float* __restrict__ ea,
                            const float* __restrict__ u,  const float* __restrict__ lu,
                            const int* __restrict__ row,  const int* __restrict__ col,
                            const int* __restrict__ eidx,
                            unsigned short* __restrict__ out, int n_e) {
  long long t = (long long)blockIdx.x * blockDim.x + threadIdx.x;
  if (t >= (long long)n_e * 800) return;
  int e = (int)(t / 800), k = (int)(t % 800);
  float v = 0.f;
  if      (k < 128) v = h [(size_t)row[e] * 128 + k];
  else if (k < 256) v = lv[(size_t)row[e] * 128 + (k - 128)];
  else if (k < 384) v = h [(size_t)col[e] * 128 + (k - 256)];
  else if (k < 512) v = lv[(size_t)col[e] * 128 + (k - 384)];
  else if (k < 514) v = ea[(size_t)e * 2 + (k - 512)];
  else if (k < 642) v = le[(size_t)e * 128 + (k - 514)];
  else if (k < 646) v = u [(size_t)eidx[e] * 4 + (k - 642)];
  else if (k < 774) v = lu[(size_t)eidx[e] * 128 + (k - 646)];
  out[t] = f2bf(v);
}

// --- v_cat = [h(128)|lv(128)|agg_e(128)|u[vi](4)|lu[vi](128)] pad 544, bf16
__global__ void gather_vcat(const float* __restrict__ h,  const float* __restrict__ lv,
                            const float* __restrict__ agg,const float* __restrict__ u,
                            const float* __restrict__ lu, const int* __restrict__ vidx,
                            unsigned short* __restrict__ out, int n_v) {
  long long t = (long long)blockIdx.x * blockDim.x + threadIdx.x;
  if (t >= (long long)n_v * 544) return;
  int i = (int)(t / 544), k = (int)(t % 544);
  float v = 0.f;
  if      (k < 128) v = h  [(size_t)i * 128 + k];
  else if (k < 256) v = lv [(size_t)i * 128 + (k - 128)];
  else if (k < 384) v = agg[(size_t)i * 128 + (k - 256)];
  else if (k < 388) v = u  [(size_t)vidx[i] * 4 + (k - 384)];
  else if (k < 516) v = lu [(size_t)vidx[i] * 128 + (k - 388)];
  out[t] = f2bf(v);
}

// --- u_cat = [u(4)|lu(128)|mean_v(128)|mean_e(128)] pad 416, bf16
__global__ void gather_ucat(const float* __restrict__ u,    const float* __restrict__ lu,
                            const float* __restrict__ sumv, const float* __restrict__ cntv,
                            const float* __restrict__ sume, const float* __restrict__ cnte,
                            unsigned short* __restrict__ out, int n_u) {
  int t = blockIdx.x * blockDim.x + threadIdx.x;
  if (t >= n_u * 416) return;
  int j = t / 416, k = t % 416;
  float v = 0.f;
  if      (k <   4) v = u [(size_t)j * 4 + k];
  else if (k < 132) v = lu[(size_t)j * 128 + (k - 4)];
  else if (k < 260) v = sumv[(size_t)j * 128 + (k - 132)] / fmaxf(cntv[j], 1.f);
  else if (k < 388) v = sume[(size_t)j * 128 + (k - 260)] / fmaxf(cnte[j], 1.f);
  out[t] = f2bf(v);
}

__global__ void cvt_f32_bf16(const float* __restrict__ in,
                             unsigned short* __restrict__ out, long long n) {
  long long t = (long long)blockIdx.x * blockDim.x + threadIdx.x;
  if (t < n) out[t] = f2bf(in[t]);
}

__global__ void scatter_add_feat(const float* __restrict__ src,
                                 const int* __restrict__ idx,
                                 float* __restrict__ dst, int n) {
  long long t = (long long)blockIdx.x * blockDim.x + threadIdx.x;
  if (t >= (long long)n * 128) return;
  int i = (int)(t >> 7);
  atomicAdd(dst + (size_t)idx[i] * 128 + (t & 127), src[t]);
}

__global__ void scatter_count(const int* __restrict__ idx, float* __restrict__ cnt, int n) {
  int i = blockIdx.x * blockDim.x + threadIdx.x;
  if (i < n) atomicAdd(cnt + idx[i], 1.f);
}

__global__ void ctx_scatter(const float* __restrict__ src, const float* __restrict__ x,
                            const int* __restrict__ vidx, float* __restrict__ dst, int n) {
  long long t = (long long)blockIdx.x * blockDim.x + threadIdx.x;
  if (t >= (long long)n * 128) return;
  int i = (int)(t >> 7);
  if (x[(size_t)i * 11] == 0.f)
    atomicAdd(dst + (size_t)vidx[i] * 128 + (t & 127), src[t]);
}

__global__ void ctx_count(const float* __restrict__ x, const int* __restrict__ vidx,
                          float* __restrict__ cnt, int n) {
  int i = blockIdx.x * blockDim.x + threadIdx.x;
  if (i < n && x[(size_t)i * 11] == 0.f) atomicAdd(cnt + vidx[i], 1.f);
}

// --- v head: out[i,:2] = [lv[i] | ctx[vidx[i]]] @ Wv(256,2) + bv
__global__ void out_v_head(const float* __restrict__ lv, const float* __restrict__ ctxs,
                           const float* __restrict__ ctxc, const int* __restrict__ vidx,
                           const float* __restrict__ Wv, const float* __restrict__ bv,
                           float* __restrict__ out, int n) {
  int i = blockIdx.x * blockDim.x + threadIdx.x;
  if (i >= n) return;
  int vi = vidx[i];
  float invc = 1.f / fmaxf(ctxc[vi], 1.f);
  float c0 = bv[0], c1 = bv[1];
  const float* a = lv + (size_t)i * 128;
  const float* s = ctxs + (size_t)vi * 128;
#pragma unroll 4
  for (int k = 0; k < 128; ++k) { c0 += a[k] * Wv[k * 2];  c1 += a[k] * Wv[k * 2 + 1]; }
#pragma unroll 4
  for (int k = 0; k < 128; ++k) {
    float v = s[k] * invc;
    c0 += v * Wv[(128 + k) * 2];  c1 += v * Wv[(128 + k) * 2 + 1];
  }
  out[(size_t)i * 2] = c0; out[(size_t)i * 2 + 1] = c1;
}

// --- scalar heads: out[i] = src[i,:] @ w(128,1) + b
__global__ void out_row_dot(const float* __restrict__ src, const float* __restrict__ w,
                            const float* __restrict__ b, float* __restrict__ out, int n) {
  int i = blockIdx.x * blockDim.x + threadIdx.x;
  if (i >= n) return;
  float c = b[0];
  const float* a = src + (size_t)i * 128;
#pragma unroll 4
  for (int k = 0; k < 128; ++k) c += a[k] * w[k];
  out[i] = c;
}

// ---------------------------------------------------------------------------
extern "C" void kernel_launch(void* const* d_in, const int* in_sizes, int n_in,
                              void* d_out, int out_size, void* d_ws, size_t ws_size,
                              hipStream_t stream) {
  (void)in_sizes; (void)n_in; (void)out_size; (void)ws_size;
  constexpr int NV = 50000, NE = 150000, NU = 32;

  // params flattened as a JAX pytree (dict keys sorted, DFS; {"b","w"}->b,w;
  // {"layers","ln"}->layers then ln{b,g}):
  // 0-3 clause_enc(b0,w0,b1,w1) 4-9 core_edge(b0,w0,b1,w1,ln.b,ln.g)
  // 10-15 core_global 16-21 core_node 22-27 dec_edge 28-33 dec_global
  // 34-39 dec_node 40-41 out_e(b,w) 42-43 out_u 44-45 out_v 46-49 var_enc
  // 50 x, 51 edge_index, 52 edge_attr, 53 u, 54 v_indices, 55 e_indices
  const float* x  = (const float*)d_in[50];
  const int*   ei = (const int*)d_in[51];
  const int*   row = ei, *col = ei + NE;
  const float* ea = (const float*)d_in[52];
  const float* uu = (const float*)d_in[53];
  const int*   vidx = (const int*)d_in[54];
  const int*   eidx = (const int*)d_in[55];
  float* out = (float*)d_out;   // v:[0,100000) e:[100000,250000) u:[250000,250032)

  // ---- workspace carve ----
  char* base = (char*)d_ws; size_t off = 0;
  auto carve = [&](size_t bytes) -> char* {
    char* p = base + off; off += (bytes + 255) & ~(size_t)255; return p;
  };
  float* h    = (float*)carve((size_t)NV * 128 * 4);
  float* lv   = (float*)carve((size_t)NV * 128 * 4);
  float* agg  = (float*)carve((size_t)NV * 128 * 4);
  float* le   = (float*)carve((size_t)NE * 128 * 4);
  float* lu   = (float*)carve((size_t)NU * 128 * 4);
  float* encv = (float*)carve((size_t)NV * 128 * 4);
  float* encc = (float*)carve((size_t)NV * 128 * 4);
  unsigned short* ecat = (unsigned short*)carve((size_t)NE * 800 * 2);
  unsigned short* vcat = (unsigned short*)carve((size_t)NV * 544 * 2);
  unsigned short* ucat = (unsigned short*)carve((size_t)NU * 416 * 2);
  unsigned short* tmpH = (unsigned short*)carve((size_t)NE * 128 * 2);
  unsigned short* cvtb = (unsigned short*)carve((size_t)NE * 128 * 2);
  unsigned short* xv   = (unsigned short*)carve((size_t)NV * 32 * 2);
  unsigned short* xc   = (unsigned short*)carve((size_t)NV * 32 * 2);
  float* sumv = (float*)carve(NU * 128 * 4);
  float* sume = (float*)carve(NU * 128 * 4);
  float* cntv = (float*)carve(NU * 4);
  float* cnte = (float*)carve(NU * 4);
  float* ctxs = (float*)carve(NU * 128 * 4);
  float* ctxc = (float*)carve(NU * 4);
  auto carveW = [&](int kpad) { return (unsigned*)carve((size_t)kpad * 64 * 4); };
  unsigned *swVar0 = carveW(32),  *swVar1 = carveW(128);
  unsigned *swCl0  = carveW(32),  *swCl1  = carveW(128);
  unsigned *swCe0  = carveW(800), *swCe1  = carveW(128);
  unsigned *swCn0  = carveW(544), *swCn1  = carveW(128);
  unsigned *swCg0  = carveW(416), *swCg1  = carveW(128);
  unsigned *swDe0  = carveW(128), *swDe1  = carveW(128);
  unsigned *swDn0  = carveW(128), *swDn1  = carveW(128);
  unsigned *swDg0  = carveW(128), *swDg1  = carveW(128);

  auto G = [](long long n) { return dim3((unsigned)((n + 255) / 256)); };
  dim3 B(256);
  auto swz = [&](int widx, unsigned* dst, int kin, int kpad) {
    int total = kpad * 64;
    swizzle_weights<<<G(total), B, 0, stream>>>((const float*)d_in[widx], dst, kin, total);
  };
  auto gemm = [&](const unsigned short* A, const unsigned* Wsw, int bidx,
                  float* oF, unsigned short* oH, int M, int K, int relu) {
    gemm_bf16_n128<<<dim3((M + 15) / 16), dim3(128), 0, stream>>>(
        A, Wsw, (const float*)d_in[bidx], oF, oH, M, K, relu);
  };
  auto ln = [&](float* X, int gidx, int bidx, int M) {
    ln_rows<<<dim3((M + 7) / 8), B, 0, stream>>>(
        X, (const float*)d_in[gidx], (const float*)d_in[bidx], M);
  };
  auto zero = [&](void* p, size_t bytes) { hipMemsetAsync(p, 0, bytes, stream); };

  // ---- swizzle all weights to wave32 WMMA B layout (bf16, K-padded) ----
  swz(47, swVar0, 10, 32);   swz(49, swVar1, 128, 128);   // var_enc
  swz(1,  swCl0,  6, 32);    swz(3,  swCl1,  128, 128);   // clause_enc
  swz(5,  swCe0,  774, 800); swz(7,  swCe1,  128, 128);   // core_edge
  swz(17, swCn0,  516, 544); swz(19, swCn1,  128, 128);   // core_node
  swz(11, swCg0,  388, 416); swz(13, swCg1,  128, 128);   // core_global
  swz(23, swDe0,  128, 128); swz(25, swDe1,  128, 128);   // dec_edge
  swz(35, swDn0,  128, 128); swz(37, swDn1,  128, 128);   // dec_node
  swz(29, swDg0,  128, 128); swz(31, swDg1,  128, 128);   // dec_global

  // ---- hetero encoder ----
  gather_xenc<<<G((long long)NV * 32), B, 0, stream>>>(x, xv, xc, NV);
  gemm(xv, swVar0, 46, nullptr, tmpH, NV, 32, 1);
  gemm(tmpH, swVar1, 48, encv, nullptr, NV, 128, 0);
  gemm(xc, swCl0, 0, nullptr, tmpH, NV, 32, 1);
  gemm(tmpH, swCl1, 2, encc, nullptr, NV, 128, 0);
  select_h<<<G((long long)NV * 128), B, 0, stream>>>(x, encv, encc, h, NV);

  zero(lv, (size_t)NV * 128 * 4);
  zero(le, (size_t)NE * 128 * 4);
  zero(lu, (size_t)NU * 128 * 4);

  // ---- 2 message-passing steps ----
  for (int step = 0; step < 2; ++step) {
    // edge block
    gather_ecat<<<G((long long)NE * 800), B, 0, stream>>>(h, lv, le, ea, uu, lu,
                                                          row, col, eidx, ecat, NE);
    gemm(ecat, swCe0, 4, nullptr, tmpH, NE, 800, 1);
    gemm(tmpH, swCe1, 6, le, nullptr, NE, 128, 1);
    ln(le, 9, 8, NE);
    // aggregate edges -> nodes (segment_sum over col)
    zero(agg, (size_t)NV * 128 * 4);
    scatter_add_feat<<<G((long long)NE * 128), B, 0, stream>>>(le, col, agg, NE);
    // node block
    gather_vcat<<<G((long long)NV * 544), B, 0, stream>>>(h, lv, agg, uu, lu, vidx, vcat, NV);
    gemm(vcat, swCn0, 16, nullptr, tmpH, NV, 544, 1);
    gemm(tmpH, swCn1, 18, lv, nullptr, NV, 128, 1);
    ln(lv, 21, 20, NV);
    // global block (segment means; gather_ucat reads lu BEFORE overwrite)
    zero(sumv, NU * 128 * 4); zero(sume, NU * 128 * 4);
    zero(cntv, NU * 4);       zero(cnte, NU * 4);
    scatter_add_feat<<<G((long long)NV * 128), B, 0, stream>>>(lv, vidx, sumv, NV);
    scatter_count<<<G(NV), B, 0, stream>>>(vidx, cntv, NV);
    scatter_add_feat<<<G((long long)NE * 128), B, 0, stream>>>(le, eidx, sume, NE);
    scatter_count<<<G(NE), B, 0, stream>>>(eidx, cnte, NE);
    gather_ucat<<<G(NU * 416), B, 0, stream>>>(uu, lu, sumv, cntv, sume, cnte, ucat, NU);
    gemm(ucat, swCg0, 10, nullptr, tmpH, NU, 416, 1);
    gemm(tmpH, swCg1, 12, lu, nullptr, NU, 128, 1);
    ln(lu, 15, 14, NU);
  }

  // ---- independent decoders ----
  cvt_f32_bf16<<<G((long long)NE * 128), B, 0, stream>>>(le, cvtb, (long long)NE * 128);
  gemm(cvtb, swDe0, 22, nullptr, tmpH, NE, 128, 1);
  gemm(tmpH, swDe1, 24, le, nullptr, NE, 128, 1);
  ln(le, 27, 26, NE);

  cvt_f32_bf16<<<G((long long)NV * 128), B, 0, stream>>>(lv, cvtb, (long long)NV * 128);
  gemm(cvtb, swDn0, 34, nullptr, tmpH, NV, 128, 1);
  gemm(tmpH, swDn1, 36, lv, nullptr, NV, 128, 1);
  ln(lv, 39, 38, NV);

  cvt_f32_bf16<<<G(NU * 128), B, 0, stream>>>(lu, cvtb, (long long)NU * 128);
  gemm(cvtb, swDg0, 28, nullptr, tmpH, NU, 128, 1);
  gemm(tmpH, swDg1, 30, lu, nullptr, NU, 128, 1);
  ln(lu, 33, 32, NU);

  // ---- global context over variable nodes + output heads ----
  zero(ctxs, NU * 128 * 4); zero(ctxc, NU * 4);
  ctx_scatter<<<G((long long)NV * 128), B, 0, stream>>>(lv, x, vidx, ctxs, NV);
  ctx_count<<<G(NV), B, 0, stream>>>(x, vidx, ctxc, NV);
  out_v_head<<<G(NV), B, 0, stream>>>(lv, ctxs, ctxc, vidx,
                                      (const float*)d_in[45], (const float*)d_in[44],
                                      out, NV);
  out_row_dot<<<G(NE), B, 0, stream>>>(le, (const float*)d_in[41],
                                       (const float*)d_in[40], out + 100000, NE);
  out_row_dot<<<G(NU), B, 0, stream>>>(lu, (const float*)d_in[43],
                                       (const float*)d_in[42], out + 250000, NU);
}
